// VariationalGCNEncoder_17918603558985
// MI455X (gfx1250) — matrix operand order repaired and checked by reading, not compiled, not run on hardware
//
#include <hip/hip_runtime.h>

typedef __attribute__((ext_vector_type(16))) __bf16 v16bf;
typedef __attribute__((ext_vector_type(8)))  __bf16 v8bf;
typedef __attribute__((ext_vector_type(4)))  __bf16 v4bf;
typedef __attribute__((ext_vector_type(8)))  float  v8f;

// ---------------------------------------------------------------------------
// Utility kernels
// ---------------------------------------------------------------------------
__global__ void k_zero_f32(float* __restrict__ p, int n) {
    int t = blockIdx.x * blockDim.x + threadIdx.x;
    if (t < n) p[t] = 0.0f;
}

__global__ void k_deg_accum(const long long* __restrict__ dst,
                            float* __restrict__ deg, long long E) {
    long long t = (long long)blockIdx.x * blockDim.x + threadIdx.x;
    if (t < E) atomicAdd(&deg[dst[t]], 1.0f);
}

__global__ void k_dis(float* __restrict__ dis, int n) {
    int t = blockIdx.x * blockDim.x + threadIdx.x;
    if (t < n) dis[t] = rsqrtf(dis[t] + 1.0f);
}

// f32 -> (hi, lo) bf16 split, 4 elements per thread (streaming, vectorized)
__global__ void k_cvt_hilo(const float* __restrict__ x, __bf16* __restrict__ hi,
                           __bf16* __restrict__ lo, int n4) {
    int t = blockIdx.x * blockDim.x + threadIdx.x;
    if (t >= n4) return;
    float4 v = ((const float4*)x)[t];
    __bf16 h0 = (__bf16)v.x, h1 = (__bf16)v.y, h2 = (__bf16)v.z, h3 = (__bf16)v.w;
    v4bf hv = {h0, h1, h2, h3};
    v4bf lv = {(__bf16)(v.x - (float)h0), (__bf16)(v.y - (float)h1),
               (__bf16)(v.z - (float)h2), (__bf16)(v.w - (float)h3)};
    ((v4bf*)hi)[t] = hv;
    ((v4bf*)lo)[t] = lv;
}

// W[Fin,Fout] f32 -> transposed hi/lo bf16 [Fout,Fin] (tiny, once per layer)
__global__ void k_cvt_w_t(const float* __restrict__ W, __bf16* __restrict__ whi,
                          __bf16* __restrict__ wlo, int Fin, int fshift) {
    int t = blockIdx.x * blockDim.x + threadIdx.x;
    int total = Fin << fshift;
    if (t >= total) return;
    int k = t >> fshift;                 // row of W
    int n = t & ((1 << fshift) - 1);     // column of W
    float v = W[t];
    __bf16 h = (__bf16)v;
    whi[(size_t)n * Fin + k] = h;
    wlo[(size_t)n * Fin + k] = (__bf16)(v - (float)h);
}

// AGG[i,f] = H[i,f] * dis[i]^2 + b[f]   (self-loop contribution + bias)
__global__ void k_init_agg(const float* __restrict__ H, const float* __restrict__ dis,
                           const float* __restrict__ b, float* __restrict__ AGG,
                           int total, int fshift) {
    int t = blockIdx.x * blockDim.x + threadIdx.x;
    if (t >= total) return;
    int i = t >> fshift;
    int f = t & ((1 << fshift) - 1);
    float d = dis[i];
    AGG[t] = H[t] * d * d + b[f];
}

// AGG[dst] += H[src] * (dis[src]*dis[dst])  -- one thread per (edge, 4-float chunk)
__global__ void k_edge_agg(const float* __restrict__ H, const float* __restrict__ dis,
                           const long long* __restrict__ src, const long long* __restrict__ dst,
                           float* __restrict__ AGG, long long E, int F, int cshift) {
    long long t = (long long)blockIdx.x * blockDim.x + threadIdx.x;
    long long e = t >> cshift;
    int c = (int)(t & ((1LL << cshift) - 1));
    if (e >= E) return;
    long long s = src[e];
    long long d = dst[e];
    float norm = dis[s] * dis[d];
    const float4 h = *((const float4*)(H + s * (long long)F) + c);
    float* out = AGG + d * (long long)F + (long long)c * 4;
    atomicAdd(out + 0, h.x * norm);
    atomicAdd(out + 1, h.y * norm);
    atomicAdd(out + 2, h.z * norm);
    atomicAdd(out + 3, h.w * norm);
}

__global__ void k_relu_inplace(float* __restrict__ p, int n) {
    int t = blockIdx.x * blockDim.x + threadIdx.x;
    if (t < n) p[t] = fmaxf(p[t], 0.0f);
}

// ---------------------------------------------------------------------------
// GEMM: H[N,Fout] = X[N,FIN] @ W[FIN,Fout] from pre-split bf16 hi/lo operands.
// One wave32 per 16x16 output tile; fully unrolled K (FIN=128 -> 12 straight-
// line v_wmma_f32_16x16x32_bf16, fed by contiguous b128 loads, zero cvt VALU).
// bf16x3: hi*hi + hi*lo + lo*hi with f32 accumulate (~fp32 accuracy).
// ---------------------------------------------------------------------------
template<int FIN>
__global__ __launch_bounds__(128)
void k_gemm_wmma(const __bf16* __restrict__ Xhi, const __bf16* __restrict__ Xlo,
                 const __bf16* __restrict__ Wthi, const __bf16* __restrict__ Wtlo,
                 float* __restrict__ H, int N, int Fout) {
    const int wave   = (blockIdx.x * blockDim.x + threadIdx.x) >> 5;
    const int lane   = threadIdx.x & 31;
    const int tilesM = (N + 15) >> 4;
    const int tilesN = Fout >> 4;
    const int tm     = wave / tilesN;
    const int tn     = wave - tm * tilesN;
    if (tm >= tilesM) return;                 // wave-uniform: EXEC stays all-1s

    const int halfsel = lane >> 4;            // 0: lanes 0-15, 1: lanes 16-31
    const int mr      = lane & 15;
    const int m       = tm * 16 + mr;         // A row for this lane
    const int n       = tn * 16 + mr;         // B/D column for this lane

    // A 16-bit layout: lane holds K = kb..kb+7 and kb+16..kb+23, kb = 8*halfsel
    const __bf16* arow_hi = Xhi  + (size_t)m * FIN + halfsel * 8;
    const __bf16* arow_lo = Xlo  + (size_t)m * FIN + halfsel * 8;
    // B 16-bit layout: lane holds column n, K = kb..kb+7 and kb+16..kb+23, kb = 8*halfsel
    const __bf16* brow_hi = Wthi + (size_t)n * FIN + halfsel * 8;
    const __bf16* brow_lo = Wtlo + (size_t)n * FIN + halfsel * 8;

    union Frag { v16bf v; v8bf h[2]; };
    v8f acc = {};
#pragma unroll
    for (int k0 = 0; k0 < FIN; k0 += 32) {
        Frag ahi, alo, bhi, blo;
        ahi.h[0] = *(const v8bf*)(arow_hi + k0);
        ahi.h[1] = *(const v8bf*)(arow_hi + k0 + 16);
        alo.h[0] = *(const v8bf*)(arow_lo + k0);
        alo.h[1] = *(const v8bf*)(arow_lo + k0 + 16);
        bhi.h[0] = *(const v8bf*)(brow_hi + k0);
        bhi.h[1] = *(const v8bf*)(brow_hi + k0 + 16);
        blo.h[0] = *(const v8bf*)(brow_lo + k0);
        blo.h[1] = *(const v8bf*)(brow_lo + k0 + 16);
        acc = __builtin_amdgcn_wmma_f32_16x16x32_bf16(false, ahi.v, false, bhi.v,
                                                      (short)0, acc, false, false);
        acc = __builtin_amdgcn_wmma_f32_16x16x32_bf16(false, ahi.v, false, blo.v,
                                                      (short)0, acc, false, false);
        acc = __builtin_amdgcn_wmma_f32_16x16x32_bf16(false, alo.v, false, bhi.v,
                                                      (short)0, acc, false, false);
    }

    // D layout: VGPR v holds row M = 8*halfsel + v, column n
    const int mrow0 = tm * 16 + halfsel * 8;
#pragma unroll
    for (int v = 0; v < 8; ++v) {
        H[(size_t)(mrow0 + v) * Fout + n] = acc[v];
    }
}

// ---------------------------------------------------------------------------
// Host-side orchestration
// ---------------------------------------------------------------------------
static inline int ceil_div_ll(long long a, int b) { return (int)((a + b - 1) / b); }

extern "C" void kernel_launch(void* const* d_in, const int* in_sizes, int n_in,
                              void* d_out, int out_size, void* d_ws, size_t ws_size,
                              hipStream_t stream) {
    const float*     x   = (const float*)d_in[0];
    const long long* ei  = (const long long*)d_in[1];   // int64 [2, E]
    const float*     W1  = (const float*)d_in[2];
    const float*     b1  = (const float*)d_in[3];
    const float*     W2  = (const float*)d_in[4];
    const float*     b2  = (const float*)d_in[5];
    const float*     W3  = (const float*)d_in[6];
    const float*     b3  = (const float*)d_in[7];
    const float*     Wmu = (const float*)d_in[8];
    const float*     bmu = (const float*)d_in[9];
    const float*     Wls = (const float*)d_in[10];
    const float*     bls = (const float*)d_in[11];

    const int Fin = 128;
    const int N   = in_sizes[0] / Fin;
    const long long E = (long long)in_sizes[1] / 2;
    const long long* src = ei;
    const long long* dst = ei + E;

    // Workspace layout: dis | H | AGG | Xhi | Xlo | Wthi | Wtlo
    char*   w   = (char*)d_ws;
    float*  dis = (float*)w;
    size_t  o   = (((size_t)N * 4) + 255) & ~(size_t)255;
    float*  H   = (float*)(w + o);  o += (size_t)N * Fin * 4;
    float*  AGG = (float*)(w + o);  o += (size_t)N * Fin * 4;
    __bf16* Xhi = (__bf16*)(w + o); o += (size_t)N * Fin * 2;
    __bf16* Xlo = (__bf16*)(w + o); o += (size_t)N * Fin * 2;
    __bf16* Wthi = (__bf16*)(w + o); o += (size_t)Fin * Fin * 2;
    __bf16* Wtlo = (__bf16*)(w + o);

    float* mu_out = (float*)d_out;                       // [N, 64]
    float* ls_out = mu_out + (size_t)N * 64;             // [N, 64]

    const int B = 256;

    // ---- degree / symmetric normalization --------------------------------
    k_zero_f32<<<ceil_div_ll(N, B), B, 0, stream>>>(dis, N);
    k_deg_accum<<<ceil_div_ll(E, B), B, 0, stream>>>(dst, dis, E);
    k_dis<<<ceil_div_ll(N, B), B, 0, stream>>>(dis, N);

    // ---- one GCN layer ----------------------------------------------------
    // cvt_x: split input activations once per layer (skipped when the hi/lo
    // split of the same activations is already resident, e.g. logstd head).
    auto gcn_layer = [&](const float* Xin, const float* Wt, const float* bt,
                         int FoL, int fshift, float* outbuf, bool relu, bool cvt_x) {
        if (cvt_x)
            k_cvt_hilo<<<ceil_div_ll((long long)N * Fin / 4, B), B, 0, stream>>>(
                Xin, Xhi, Xlo, N * Fin / 4);
        k_cvt_w_t<<<ceil_div_ll((long long)Fin << fshift, B), B, 0, stream>>>(
            Wt, Wthi, Wtlo, Fin, fshift);
        int waves  = ((N + 15) / 16) * (FoL / 16);
        int blocks = (waves * 32 + 127) / 128;
        k_gemm_wmma<128><<<blocks, 128, 0, stream>>>(Xhi, Xlo, Wthi, Wtlo, H, N, FoL);
        k_init_agg<<<ceil_div_ll((long long)N << fshift, B), B, 0, stream>>>(
            H, dis, bt, outbuf, N << fshift, fshift);
        int cshift = fshift - 2;  // float4 chunks per row
        k_edge_agg<<<ceil_div_ll(E << cshift, B), B, 0, stream>>>(
            H, dis, src, dst, outbuf, E, FoL, cshift);
        if (relu)
            k_relu_inplace<<<ceil_div_ll((long long)N << fshift, B), B, 0, stream>>>(
                outbuf, N << fshift);
    };

    // Hidden layers: activations cycle through AGG (relu in place); stream
    // ordering guarantees each gemm reads AGG before init_agg rewrites it.
    gcn_layer(x,   W1, b1, 128, 7, AGG, true, true);
    gcn_layer(AGG, W2, b2, 128, 7, AGG, true, true);
    gcn_layer(AGG, W3, b3, 128, 7, AGG, true, true);

    // Output heads write straight into d_out (no relu). AGG (= h) is left
    // intact by the mu head, so logstd reuses both AGG and its bf16 split.
    gcn_layer(AGG, Wmu, bmu, 64, 6, mu_out, false, true);
    gcn_layer(AGG, Wls, bls, 64, 6, ls_out, false, false);
}
